// DiffAttention_78786880078292
// MI455X (gfx1250) — matrix operand order repaired and checked by reading, not compiled, not run on hardware
//
#include <hip/hip_runtime.h>
#include <hip/hip_bf16.h>

typedef __attribute__((ext_vector_type(16))) _Float16 v16h;
typedef __attribute__((ext_vector_type(8)))  _Float16 v8h;
typedef __attribute__((ext_vector_type(8)))  float    v8f;

union AFrag { v16h v; v8h h[2]; };

#define WMMA_F16(A, Bf, C) \
  __builtin_amdgcn_wmma_f32_16x16x32_f16(false, (A), false, (Bf), (short)0, (C), false, false)

// ---------------------------------------------------------------------------
// fp32 -> fp16 convert
// ---------------------------------------------------------------------------
__global__ __launch_bounds__(256) void cvt_f32_to_f16(const float* __restrict__ in,
                                                      _Float16* __restrict__ out,
                                                      size_t n) {
  size_t i = (size_t)blockIdx.x * blockDim.x + threadIdx.x;
  if (i < n) out[i] = (_Float16)in[i];
}

// ---------------------------------------------------------------------------
// lambda_full = exp(sum(q1*k1)) - exp(sum(q2*k2)) + LAMBDA_INIT   (D = 64)
// ---------------------------------------------------------------------------
__global__ __launch_bounds__(64) void lambda_kernel(const float* __restrict__ q1,
                                                    const float* __restrict__ q2,
                                                    const float* __restrict__ k1,
                                                    const float* __restrict__ k2,
                                                    float* __restrict__ out) {
  __shared__ float a[64], c[64];
  int t = threadIdx.x;
  a[t] = q1[t] * k1[t];
  c[t] = q2[t] * k2[t];
  __syncthreads();
  for (int off = 32; off > 0; off >>= 1) {
    if (t < off) { a[t] += a[t + off]; c[t] += c[t + off]; }
    __syncthreads();
  }
  if (t == 0) {
    const float LAMBDA_INIT = 0.8f - 0.6f * __expf(-3.6f);
    out[0] = __expf(a[0]) - __expf(c[0]) + LAMBDA_INIT;
  }
}

// ---------------------------------------------------------------------------
// GEMM: C[M,N] = A[M,K] (f16, row-major) * W[N,K]^T (f16, row-major), f32 out
// Block = 256 threads = 8 waves; each wave computes a 16(M) x 64(N) strip.
// B stage = 64(N) x 64(K) f16 = 8KB, shared by all waves, double-buffered in
// LDS via GLOBAL_LOAD_ASYNC_TO_LDS_B128 (each thread copies 32B = 2 ops).
// Steady-state loop is branch-free: wait<=2; barrier; 8 WMMA; barrier; issue.
// All 8 B-fragments are loaded before the WMMA burst so the DS loads can be
// clause'd and drained with incremental s_wait_dscnt.
// A-frag: lane l -> row m0+(l&15); halves = A[m][k + 16*(l>>4) + 0..15]
// B-frag: lane l -> col n        ; halves = Btile[n][j + 16*(l>>4) + 0..15]
// ---------------------------------------------------------------------------
__global__ __launch_bounds__(256)
void gemm_f16f32(const _Float16* __restrict__ A, const _Float16* __restrict__ W,
                 float* __restrict__ C, int M, int N, int K) {
  __shared__ __align__(128) _Float16 ldsB[2][64][64];  // 2 x 8KB

  const int tid  = threadIdx.x;
  const int lane = tid & 31;
  const int wave = tid >> 5;
  const int m0 = blockIdx.y * 128 + wave * 16;
  const int n0 = blockIdx.x * 64;
  const int lm = lane & 15;
  const int hi = lane >> 4;
  const int ka = hi * 16;
  const int kb = hi * 16;

  // async-copy assignment: thread t moves 32B: tile row t>>2, halves (t&3)*16
  const int brow = tid >> 2;
  const int bcol = (tid & 3) * 16;
  const _Float16* gsrc = W + (size_t)(n0 + brow) * K + bcol;
  const unsigned ldsa[2] = {
      (unsigned)(uintptr_t)&ldsB[0][brow][bcol],
      (unsigned)(uintptr_t)&ldsB[1][brow][bcol]};

  v8f acc[4];
#pragma unroll
  for (int t = 0; t < 4; ++t) acc[t] = {};

  const _Float16* arow = A + (size_t)(m0 + lm) * K;
  const int nstage = K / 64;  // >= 2 for all uses here

  auto issue = [&](int kk, int buf) {
    const _Float16* g = gsrc + kk;
    asm volatile("global_load_async_to_lds_b128 %0, %1, off"
                 :: "v"(ldsa[buf]), "v"(g) : "memory");
    asm volatile("global_load_async_to_lds_b128 %0, %1, off"
                 :: "v"(ldsa[buf] + 16), "v"(g + 8) : "memory");
  };

  auto compute = [&](int k, int buf) {
    AFrag a0, a1;
    a0.h[0] = *(const v8h*)(arow + k + ka);
    a0.h[1] = *(const v8h*)(arow + k + ka + 8);
    a1.h[0] = *(const v8h*)(arow + k + 32 + ka);
    a1.h[1] = *(const v8h*)(arow + k + 32 + ka + 8);
    v16h bf[8];
#pragma unroll
    for (int t = 0; t < 4; ++t) {
      bf[2 * t]     = *(const v16h*)(&ldsB[buf][t * 16 + lm][kb]);
      bf[2 * t + 1] = *(const v16h*)(&ldsB[buf][t * 16 + lm][32 + kb]);
    }
#pragma unroll
    for (int t = 0; t < 4; ++t) {
      acc[t] = WMMA_F16(a0.v, bf[2 * t], acc[t]);
      acc[t] = WMMA_F16(a1.v, bf[2 * t + 1], acc[t]);
    }
  };

  // prologue: stages 0 and 1 in flight
  issue(0, 0);
  issue(64, 1);

  int i = 0;
  for (; i < nstage - 2; ++i) {
    __builtin_prefetch(arow + i * 64 + 256, 0, 3);
    asm volatile("s_wait_asynccnt 0x2" ::: "memory");  // stage i landed
    __syncthreads();
    compute(i * 64, i & 1);
    __syncthreads();                                   // all readers done
    issue((i + 2) * 64, i & 1);
  }
  // tail: stage nstage-2 (next stage still in flight), then final stage
  asm volatile("s_wait_asynccnt 0x2" ::: "memory");
  __syncthreads();
  compute(i * 64, i & 1);
  ++i;
  asm volatile("s_wait_asynccnt 0x0" ::: "memory");
  __syncthreads();
  compute(i * 64, i & 1);

  float* crow = C + (size_t)(m0 + hi * 8) * N + n0 + lm;
#pragma unroll
  for (int t = 0; t < 4; ++t)
#pragma unroll
    for (int r = 0; r < 8; ++r)
      crow[(size_t)r * N + t * 16] = acc[t][r];
}

// ---------------------------------------------------------------------------
// RoPE + optional scale; repack (B,S,2H,64) f32 -> [B][2H][S][64] f16
// ---------------------------------------------------------------------------
__global__ __launch_bounds__(256)
void rope_pack(const float* __restrict__ X, const float* __restrict__ fc,
               const float* __restrict__ fs, _Float16* __restrict__ Out,
               float scale, int S) {
  size_t gid = (size_t)blockIdx.x * blockDim.x + threadIdx.x;  // B*S*32*32 pairs
  int d2 = (int)(gid & 31);
  size_t r = gid >> 5;
  int sh = (int)(r & 31); r >>= 5;
  int s = (int)(r % (size_t)S);
  int b = (int)(r / (size_t)S);
  const float* xp = X + ((size_t)(b * S + s) * 2048) + sh * 64 + 2 * d2;
  float re = xp[0], im = xp[1];
  float c = fc[s * 32 + d2], sn = fs[s * 32 + d2];
  _Float16* op = Out + (((size_t)(b * 32 + sh) * S + s) * 64) + 2 * d2;
  op[0] = (_Float16)((re * c - im * sn) * scale);
  op[1] = (_Float16)((re * sn + im * c) * scale);
}

// ---------------------------------------------------------------------------
// V transpose: (B,S,H,128) f32 -> [B][H][128][S] f16
// ---------------------------------------------------------------------------
__global__ __launch_bounds__(256)
void vtrans_kernel(const float* __restrict__ X, _Float16* __restrict__ Vt, int S) {
  size_t gid = (size_t)blockIdx.x * blockDim.x + threadIdx.x;  // B*S*16*128
  int d = (int)(gid & 127);
  size_t r = gid >> 7;
  int h = (int)(r & 15); r >>= 4;
  int s = (int)(r % (size_t)S);
  int b = (int)(r / (size_t)S);
  float v = X[((size_t)(b * S + s) * 2048) + h * 128 + d];
  Vt[(((size_t)(b * 16 + h) * 128 + d) * (size_t)S) + s] = (_Float16)v;
}

// ---------------------------------------------------------------------------
// Flash attention per sub-head, causal. grid = (S/64, 2H, B), block = 4 waves.
// Each wave owns 16 query rows; K-tile = 32 keys per iteration.
// O[b][sh][s][0..127] = softmax(q k^T) V  (already normalized by row sum)
// ---------------------------------------------------------------------------
__global__ __launch_bounds__(128)
void attn_kernel(const _Float16* __restrict__ Q, const _Float16* __restrict__ Kh,
                 const _Float16* __restrict__ Vt, float* __restrict__ O, int S) {
  __shared__ __align__(16) _Float16 lds[4][16][32];
  const int lane = threadIdx.x & 31;
  const int wave = threadIdx.x >> 5;
  const int qt = blockIdx.x;
  const int sh = blockIdx.y;
  const int b  = blockIdx.z;
  const int h  = sh >> 1;
  const int q0 = qt * 64 + wave * 16;
  const int lm = lane & 15;
  const int hi = lane >> 4;
  const int ka8 = hi * 16;
  const int k16 = hi * 16;

  const _Float16* qbase = Q  + (size_t)(b * 32 + sh) * S * 64;
  const _Float16* kbase = Kh + (size_t)(b * 32 + sh) * S * 64;
  const _Float16* vbase = Vt + (size_t)(b * 16 + h) * 128 * (size_t)S;

  // Resident Q fragments (rows q0..q0+15, d = 0..63 -> two K=32 slabs)
  const _Float16* qrow = qbase + (size_t)(q0 + lm) * 64;
  AFrag aq0, aq1;
  aq0.h[0] = *(const v8h*)(qrow + ka8);
  aq0.h[1] = *(const v8h*)(qrow + ka8 + 8);
  aq1.h[0] = *(const v8h*)(qrow + 32 + ka8);
  aq1.h[1] = *(const v8h*)(qrow + 32 + ka8 + 8);

  v8f acc[8];
#pragma unroll
  for (int t = 0; t < 8; ++t) acc[t] = {};
  float mrow[8], lrow[8];
#pragma unroll
  for (int r = 0; r < 8; ++r) { mrow[r] = -INFINITY; lrow[r] = 0.0f; }

  const int ntile = (q0 + 16 + 31) / 32;
  for (int kt = 0; kt < ntile; ++kt) {
    const int n0 = kt * 32;
    // ---- scores: 16 rows x 32 keys, K-dim = 64 ----
    v8f s0 = {}, s1 = {};
    {
      const _Float16* k0 = kbase + (size_t)(n0 + lm) * 64 + k16;
      const _Float16* k1 = kbase + (size_t)(n0 + 16 + lm) * 64 + k16;
      v16h b00 = *(const v16h*)(k0);
      v16h b01 = *(const v16h*)(k0 + 32);
      v16h b10 = *(const v16h*)(k1);
      v16h b11 = *(const v16h*)(k1 + 32);
      s0 = WMMA_F16(aq0.v, b00, s0);
      s0 = WMMA_F16(aq1.v, b01, s0);
      s1 = WMMA_F16(aq0.v, b10, s1);
      s1 = WMMA_F16(aq1.v, b11, s1);
    }
    // ---- causal mask (wave-uniform branch) ----
    if (n0 + 31 > q0) {
#pragma unroll
      for (int r = 0; r < 8; ++r) {
        int m = q0 + r + 8 * hi;
        if (n0 + lm > m)      s0[r] = -1e30f;
        if (n0 + 16 + lm > m) s1[r] = -1e30f;
      }
    }
    // ---- online softmax (row stats via width-16 butterflies) ----
    float corr[8], p0v[8], p1v[8];
#pragma unroll
    for (int r = 0; r < 8; ++r) {
      float v = fmaxf(s0[r], s1[r]);
      v = fmaxf(v, __shfl_xor(v, 1, 16));
      v = fmaxf(v, __shfl_xor(v, 2, 16));
      v = fmaxf(v, __shfl_xor(v, 4, 16));
      v = fmaxf(v, __shfl_xor(v, 8, 16));
      float nm = fmaxf(mrow[r], v);
      float c  = __expf(mrow[r] - nm);
      float p0 = __expf(s0[r] - nm);
      float p1 = __expf(s1[r] - nm);
      float ps = p0 + p1;
      ps += __shfl_xor(ps, 1, 16);
      ps += __shfl_xor(ps, 2, 16);
      ps += __shfl_xor(ps, 4, 16);
      ps += __shfl_xor(ps, 8, 16);
      lrow[r] = lrow[r] * c + ps;
      mrow[r] = nm;
      corr[r] = c; p0v[r] = p0; p1v[r] = p1;
    }
#pragma unroll
    for (int t = 0; t < 8; ++t)
#pragma unroll
      for (int r = 0; r < 8; ++r) acc[t][r] *= corr[r];

    // ---- C-layout P -> LDS -> A-layout fragment ----
#pragma unroll
    for (int r = 0; r < 8; ++r) {
      int ml = r + 8 * hi;
      lds[wave][ml][lm]      = (_Float16)p0v[r];
      lds[wave][ml][lm + 16] = (_Float16)p1v[r];
    }
    asm volatile("s_wait_dscnt 0x0" ::: "memory");
    AFrag ap;
    ap.h[0] = *(const v8h*)(&lds[wave][lm][ka8]);
    ap.h[1] = *(const v8h*)(&lds[wave][lm][ka8 + 8]);
    asm volatile("" ::: "memory");

    // ---- acc += P (16x32) * V (32x128) ----
#pragma unroll
    for (int t = 0; t < 8; ++t) {
      const _Float16* vp = vbase + (size_t)(t * 16 + lm) * (size_t)S + n0 + k16;
      v16h bv = *(const v16h*)vp;
      acc[t] = WMMA_F16(ap.v, bv, acc[t]);
    }
  }

  // ---- normalize and store ----
  float inv[8];
#pragma unroll
  for (int r = 0; r < 8; ++r) inv[r] = 1.0f / lrow[r];
  float* orow = O + ((size_t)(b * 32 + sh) * S + q0 + 8 * hi) * 128 + lm;
#pragma unroll
  for (int t = 0; t < 8; ++t)
#pragma unroll
    for (int r = 0; r < 8; ++r)
      orow[(size_t)r * 128 + t * 16] = acc[t][r] * inv[r];
}

// ---------------------------------------------------------------------------
// combine: ctx = rmsnorm(o0 - lambda*o1) * norm_w, packed to f16 (B*S) x 2048
// grid = (H, S, B), block = 128 (one value-dim each)
// ---------------------------------------------------------------------------
__global__ __launch_bounds__(128)
void combine_kernel(const float* __restrict__ O, const float* __restrict__ lam_p,
                    const float* __restrict__ norm_w, _Float16* __restrict__ ctx,
                    int S) {
  __shared__ float red[128];
  const int d = threadIdx.x;
  const int h = blockIdx.x;
  const int s = blockIdx.y;
  const int b = blockIdx.z;
  const float lam = lam_p[0];
  size_t i0 = ((size_t)(b * 32 + 2 * h)     * S + s) * 128 + d;
  size_t i1 = ((size_t)(b * 32 + 2 * h + 1) * S + s) * 128 + d;
  float v = O[i0] - lam * O[i1];
  red[d] = v * v;
  __syncthreads();
  for (int off = 64; off > 0; off >>= 1) {
    if (d < off) red[d] += red[d + off];
    __syncthreads();
  }
  float rms = rsqrtf(red[0] / 128.0f + 1e-5f);
  ctx[((size_t)(b * S + s) * 2048) + h * 128 + d] = (_Float16)(v * rms * norm_w[d]);
}

// ---------------------------------------------------------------------------
extern "C" void kernel_launch(void* const* d_in, const int* in_sizes, int n_in,
                              void* d_out, int out_size, void* d_ws, size_t ws_size,
                              hipStream_t stream) {
  (void)in_sizes; (void)n_in; (void)out_size; (void)ws_size;
  constexpr int B = 2, S = 2048, E = 2048, H = 16;
  constexpr int M = B * S;  // 4096

  const float* x   = (const float*)d_in[0];
  const float* Wq  = (const float*)d_in[1];
  const float* Wk  = (const float*)d_in[2];
  const float* Wv  = (const float*)d_in[3];
  const float* Wo  = (const float*)d_in[4];
  const float* lq1 = (const float*)d_in[5];
  const float* lq2 = (const float*)d_in[6];
  const float* lk1 = (const float*)d_in[7];
  const float* lk2 = (const float*)d_in[8];
  const float* nw  = (const float*)d_in[9];
  const float* fc  = (const float*)d_in[10];
  const float* fs  = (const float*)d_in[11];
  float* out = (float*)d_out;

  // workspace carve-up (256B aligned)
  char* ws = (char*)d_ws;
  size_t off = 0;
  auto carve = [&](size_t bytes) { size_t o = off; off += (bytes + 255) & ~(size_t)255; return o; };
  _Float16* xf16 = (_Float16*)(ws + carve((size_t)M * E * 2));
  _Float16* wf16 = (_Float16*)(ws + carve((size_t)E * E * 2));
  float*    tmp  = (float*)   (ws + carve((size_t)M * E * 4));
  _Float16* qh   = (_Float16*)(ws + carve((size_t)B * 32 * S * 64 * 2));
  _Float16* kh   = (_Float16*)(ws + carve((size_t)B * 32 * S * 64 * 2));
  _Float16* vt   = (_Float16*)(ws + carve((size_t)B * 16 * 128 * S * 2));
  float*    ovec = (float*)   (ws + carve((size_t)B * 32 * S * 128 * 4));
  _Float16* ctx  = (_Float16*)(ws + carve((size_t)M * E * 2));
  float*    lam  = (float*)   (ws + carve(256));

  const dim3 gemm_grid(E / 64, M / 128);
  const size_t nXE = (size_t)M * E;

  // lambda scalar
  lambda_kernel<<<1, 64, 0, stream>>>(lq1, lq2, lk1, lk2, lam);

  // x -> f16
  cvt_f32_to_f16<<<(int)((nXE + 255) / 256), 256, 0, stream>>>(x, xf16, nXE);

  const size_t nW = (size_t)E * E;
  const int wblocks = (int)((nW + 255) / 256);
  const int rope_blocks = (int)(((size_t)B * S * 32 * 32 + 255) / 256);
  const int vt_blocks   = (int)(((size_t)B * S * 16 * 128 + 255) / 256);

  // Q = rope(x Wq^T) * D^-1/2
  cvt_f32_to_f16<<<wblocks, 256, 0, stream>>>(Wq, wf16, nW);
  gemm_f16f32<<<gemm_grid, 256, 0, stream>>>(xf16, wf16, tmp, M, E, E);
  rope_pack<<<rope_blocks, 256, 0, stream>>>(tmp, fc, fs, qh, 0.125f, S);

  // K = rope(x Wk^T)
  cvt_f32_to_f16<<<wblocks, 256, 0, stream>>>(Wk, wf16, nW);
  gemm_f16f32<<<gemm_grid, 256, 0, stream>>>(xf16, wf16, tmp, M, E, E);
  rope_pack<<<rope_blocks, 256, 0, stream>>>(tmp, fc, fs, kh, 1.0f, S);

  // V = x Wv^T, transposed per head
  cvt_f32_to_f16<<<wblocks, 256, 0, stream>>>(Wv, wf16, nW);
  gemm_f16f32<<<gemm_grid, 256, 0, stream>>>(xf16, wf16, tmp, M, E, E);
  vtrans_kernel<<<vt_blocks, 256, 0, stream>>>(tmp, vt, S);

  // flash attention per sub-head
  attn_kernel<<<dim3(S / 64, 2 * H, B), 128, 0, stream>>>(qh, kh, vt, ovec, S);

  // differential combine + RMS norm
  combine_kernel<<<dim3(H, S, B), 128, 0, stream>>>(ovec, lam, nw, ctx, S);

  // out = ctx Wo^T
  cvt_f32_to_f16<<<wblocks, 256, 0, stream>>>(Wo, wf16, nW);
  gemm_f16f32<<<gemm_grid, 256, 0, stream>>>(ctx, wf16, out, M, E, E);
}
